// MultilayerGRU_43387759624628
// MI455X (gfx1250) — matrix operand order repaired and checked by reading, not compiled, not run on hardware
//
#include <hip/hip_runtime.h>
#include <hip/hip_bf16.h>

#define S_LEN   1024
#define BATCH   64
#define INDIM   128
#define HDIM    512
#define ODIM    128
#define NLAYER  3
#define NWG_L   8            // workgroups per layer (64 output cols each)
#define NWG_H   2            // workgroups for output head (64 cols each)
#define TOTAL_WG (NLAYER*NWG_L + NWG_H)   // 26
#define NTICKS  (S_LEN + NLAYER)          // ticks 0 .. S+2
#define XSTRIDE 136          // 128 staged cols + 8 pad halfs -> 4-bank row skew

typedef __attribute__((ext_vector_type(8)))  float  v8f;
typedef __attribute__((ext_vector_type(8)))  __bf16 v8bf;
typedef __attribute__((ext_vector_type(16))) __bf16 v16bf;

// A-fragment (16x32 bf16): row = lane&15; lanes<16 hold K {0..7,16..23}, lanes>=16 hold {8..15,24..31}
__device__ __forceinline__ v16bf load_fragA(const __bf16* base, int stride, int row, int kblk, int lane) {
    const int khalf = (lane >> 4) & 1;
    const __bf16* p = base + row * stride + kblk + khalf * 8;
    union { v16bf v; v8bf h[2]; } u;
    u.h[0] = *(const v8bf*)(p);
    u.h[1] = *(const v8bf*)(p + 16);
    return u.v;
}
// B-fragment (32x16 bf16) loaded from row-major W[N,K] (== B^T): col n = lane&15,
// lanes<16 hold K {0..7,16..23}, lanes>=16 hold {8..15,24..31}
__device__ __forceinline__ v16bf load_fragB(const __bf16* base, int stride, int row, int kblk, int lane) {
    const int khalf = (lane >> 4) & 1;
    const __bf16* p = base + row * stride + kblk + khalf * 8;
    union { v16bf v; v8bf h[2]; } u;
    u.h[0] = *(const v8bf*)(p);
    u.h[1] = *(const v8bf*)(p + 16);
    return u.v;
}

__device__ __forceinline__ float sigm(float x) { return 1.0f / (1.0f + expf(-x)); }

__device__ __forceinline__ void tick_barrier(int* bar, int k) {
    __syncthreads();
    if (threadIdx.x == 0) {
        __threadfence();
        __hip_atomic_fetch_add(&bar[k], 1, __ATOMIC_ACQ_REL, __HIP_MEMORY_SCOPE_AGENT);
        while (__hip_atomic_load(&bar[k], __ATOMIC_ACQUIRE, __HIP_MEMORY_SCOPE_AGENT) < TOTAL_WG)
            __builtin_amdgcn_s_sleep(2);
    }
    __syncthreads();
}

__global__ void f2bf_kernel(const float* __restrict__ s, __bf16* __restrict__ d, int n) {
    int i = blockIdx.x * blockDim.x + threadIdx.x;
    if (i < n) d[i] = (__bf16)s[i];
}

__global__ void hinit_kernel(const float* __restrict__ h0, float* __restrict__ hbuf) {
    int i = blockIdx.x * blockDim.x + threadIdx.x;
    if (i < BATCH * NLAYER * HDIM) {
        int b = i / (NLAYER * HDIM);
        int rem = i % (NLAYER * HDIM);
        int l = rem / HDIM;
        int n = rem % HDIM;
        // parity-1 buffer is read at t=0 (reads (t-1)&1 == 1)
        hbuf[((size_t)l * 2 + 1) * (BATCH * HDIM) + (size_t)b * HDIM + n] = h0[i];
    }
}

__global__ void __launch_bounds__(512, 1) gru_persistent(
    const float* __restrict__ inX,
    const float* __restrict__ wxz0, const float* __restrict__ bxz0,
    const float* __restrict__ whz0, const float* __restrict__ whr0,
    const float* __restrict__ bxg0,
    const float* __restrict__ wxzS, const float* __restrict__ bxzS,
    const float* __restrict__ whzS, const float* __restrict__ whrS,
    const float* __restrict__ bxgS,
    const float* __restrict__ bhy,
    const __bf16* __restrict__ wxg0b, const __bf16* __restrict__ whg0b,
    const __bf16* __restrict__ wxgSb, const __bf16* __restrict__ whgSb,
    const __bf16* __restrict__ whyb,
    float* __restrict__ hbuf, __bf16* __restrict__ xbuf,
    int* __restrict__ bar, float* __restrict__ out)
{
    __shared__ float  zArr[BATCH];
    __shared__ float  rArr[BATCH];
    __shared__ __bf16 xs [BATCH * XSTRIDE];
    __shared__ __bf16 rhs[BATCH * XSTRIDE];

    const int bid  = blockIdx.x;
    const int tid  = threadIdx.x;
    const int lane = tid & 31;
    const int wave = tid >> 5;
    const int mi   = wave >> 2;       // M tile 0..3  (rows of batch)
    const int ni   = wave & 3;        // N tile 0..3  (cols of slice)
    const bool isHead = (bid >= NLAYER * NWG_L);

    if (!isHead) {
        const int li = bid >> 3;
        const int n0 = (bid & 7) * 64;
        const int K_in = (li == 0) ? INDIM : HDIM;
        const float* wxz = (li == 0) ? wxz0 : (wxzS + (li - 1) * HDIM);
        const float  bxz = (li == 0) ? bxz0[0] : bxzS[li - 1];
        const float* whz = (li == 0) ? whz0 : (whzS + (li - 1) * HDIM);
        const float* whr = (li == 0) ? whr0 : (whrS + (li - 1) * HDIM);
        const float* bxg = (li == 0) ? bxg0 : (bxgS + (li - 1) * HDIM);
        const __bf16* wxgb = (li == 0) ? wxg0b : (wxgSb + (size_t)(li - 1) * HDIM * HDIM);
        const __bf16* whgb = (li == 0) ? whg0b : (whgSb + (size_t)(li - 1) * HDIM * HDIM);
        float*        hbase = hbuf + (size_t)li * 2 * BATCH * HDIM;
        const __bf16* xsrcB = (li == 0) ? (const __bf16*)nullptr
                                        : (xbuf + (size_t)(li - 1) * 2 * BATCH * HDIM);
        __bf16*       xdstB = xbuf + (size_t)li * 2 * BATCH * HDIM;

        for (int k = 0; k < NTICKS; ++k) {
            const int t = k - li;
            if (t >= 0 && t < S_LEN) {
                const float*  hprev = hbase + (size_t)((t + 1) & 1) * (BATCH * HDIM);
                float*        hnext = hbase + (size_t)(t & 1) * (BATCH * HDIM);
                const __bf16* xsrc  = (li == 0) ? (const __bf16*)nullptr
                                                : (xsrcB + (size_t)(t & 1) * (BATCH * HDIM));
                __bf16*       xdst  = xdstB + (size_t)(t & 1) * (BATCH * HDIM);

                // ---- phase 1: scalar gates z[b], r[b] (shuffle-reduced dot products)
                {
                    const int b = tid >> 3, c = tid & 7;
                    float sa = 0.f, sz = 0.f, sr = 0.f;
                    const float* hrow = hprev + (size_t)b * HDIM;
                    for (int kk = c * 64; kk < c * 64 + 64; ++kk) {
                        float hv = hrow[kk];
                        sz += hv * whz[kk];
                        sr += hv * whr[kk];
                    }
                    const int ck = K_in >> 3;
                    if (li == 0) {
                        const float* xrow = inX + ((size_t)b * S_LEN + t) * INDIM;
                        for (int kk = c * ck; kk < c * ck + ck; ++kk) sa += xrow[kk] * wxz[kk];
                    } else {
                        const __bf16* xrow = xsrc + (size_t)b * HDIM;
                        for (int kk = c * ck; kk < c * ck + ck; ++kk) sa += (float)xrow[kk] * wxz[kk];
                    }
                    for (int off = 4; off > 0; off >>= 1) {
                        sa += __shfl_down(sa, off, 8);
                        sz += __shfl_down(sz, off, 8);
                        sr += __shfl_down(sr, off, 8);
                    }
                    if (c == 0) {
                        float a = sa + bxz;
                        zArr[b] = sigm(a + sz);
                        rArr[b] = sigm(a + sr);
                    }
                }
                __syncthreads();

                // ---- phase 2/3: staged K-quarter GEMMs  acc = x@Wxg^T + (r*h)@Whg^T
                v8f acc = {0.f, 0.f, 0.f, 0.f, 0.f, 0.f, 0.f, 0.f};
                const int arow = mi * 16 + (lane & 15);
                const int nrow = n0 + ni * 16 + (lane & 15);

                for (int q = 0; q < 4; ++q) {
                    const int kBase = q * 128;
                    // vectorized staging: K pairs, packed bf16x2 DWORD stores
                    for (int i = tid; i < BATCH * 64; i += 512) {
                        const int b  = i >> 6;
                        const int kp = (i & 63) * 2;
                        const int kg = kBase + kp;
                        float x0, x1;
                        if (li == 0) {
                            if (kg < INDIM) {
                                const float* xr = inX + ((size_t)b * S_LEN + t) * INDIM + kg;
                                x0 = xr[0]; x1 = xr[1];
                            } else { x0 = 0.f; x1 = 0.f; }
                        } else {
                            union { unsigned u; __bf16 h[2]; } xin;
                            xin.u = *(const unsigned*)(xsrc + (size_t)b * HDIM + kg);
                            x0 = (float)xin.h[0]; x1 = (float)xin.h[1];
                        }
                        const float r  = rArr[b];
                        const float2 h2 = *(const float2*)(hprev + (size_t)b * HDIM + kg);
                        union { unsigned u; __bf16 h[2]; } ux, ur;
                        ux.h[0] = (__bf16)x0;        ux.h[1] = (__bf16)x1;
                        ur.h[0] = (__bf16)(r * h2.x); ur.h[1] = (__bf16)(r * h2.y);
                        *(unsigned*)(xs  + b * XSTRIDE + kp) = ux.u;
                        *(unsigned*)(rhs + b * XSTRIDE + kp) = ur.u;
                    }
                    __syncthreads();
                    // prefetch next quarter's weight rows while this quarter computes
                    if (q < 3) {
                        const int kn = kBase + 128;
                        __builtin_prefetch(whgb + (size_t)nrow * HDIM + kn, 0, 1);
                        if (kn < K_in)
                            __builtin_prefetch(wxgb + (size_t)nrow * K_in + kn, 0, 1);
                    }
                    for (int kk = 0; kk < 4; ++kk) {
                        const int kg = kBase + kk * 32;
                        if (kg < K_in) {
                            v16bf a  = load_fragA(xs, XSTRIDE, arow, kk * 32, lane);
                            v16bf bf = load_fragB(wxgb, K_in, nrow, kg, lane);
                            acc = __builtin_amdgcn_wmma_f32_16x16x32_bf16(
                                false, a, false, bf, (short)0, acc, false, false);
                        }
                        {
                            v16bf a  = load_fragA(rhs, XSTRIDE, arow, kk * 32, lane);
                            v16bf bf = load_fragB(whgb, HDIM, nrow, kg, lane);
                            acc = __builtin_amdgcn_wmma_f32_16x16x32_bf16(
                                false, a, false, bf, (short)0, acc, false, false);
                        }
                    }
                    __syncthreads();
                }

                // ---- phase 4: epilogue  h' = h*z + (1-z)*tanh(acc + bxg)
                {
                    const float bias = bxg[nrow];
                    const int mhi = (lane >> 4) << 3;
                    for (int p = 0; p < 8; ++p) {
                        int b = mi * 16 + mhi + p;
                        float g  = tanhf(acc[p] + bias);
                        float z  = zArr[b];
                        float hp = hprev[(size_t)b * HDIM + nrow];
                        float hn = hp * z + (1.f - z) * g;
                        hnext[(size_t)b * HDIM + nrow] = hn;
                        xdst [(size_t)b * HDIM + nrow] = (__bf16)hn;
                        if (t == S_LEN - 1)
                            out[(size_t)BATCH * S_LEN * ODIM +
                                (size_t)b * (NLAYER * HDIM) + (size_t)li * HDIM + nrow] = hn;
                    }
                }
            }
            tick_barrier(bar, k);
        }
    } else {
        // ---- output head: y[t] = x2[t] @ Why^T + bhy, pipelined at t = k-3
        const int n0 = (bid - NLAYER * NWG_L) * 64;
        const __bf16* x3Base = xbuf + (size_t)2 * 2 * BATCH * HDIM;
        const int arow = mi * 16 + (lane & 15);
        const int nrow = n0 + ni * 16 + (lane & 15);
        for (int k = 0; k < NTICKS; ++k) {
            const int t = k - NLAYER;
            if (t >= 0 && t < S_LEN) {
                const __bf16* x3 = x3Base + (size_t)(t & 1) * (BATCH * HDIM);
                v8f acc = {0.f, 0.f, 0.f, 0.f, 0.f, 0.f, 0.f, 0.f};
                for (int it = 0; it < 16; ++it) {
                    v16bf a  = load_fragA(x3, HDIM, arow, it * 32, lane);
                    v16bf bf = load_fragB(whyb, HDIM, nrow, it * 32, lane);
                    acc = __builtin_amdgcn_wmma_f32_16x16x32_bf16(
                        false, a, false, bf, (short)0, acc, false, false);
                }
                const float bias = bhy[nrow];
                const int mhi = (lane >> 4) << 3;
                for (int p = 0; p < 8; ++p) {
                    int b = mi * 16 + mhi + p;
                    out[(size_t)b * S_LEN * ODIM + (size_t)t * ODIM + nrow] = acc[p] + bias;
                }
            }
            tick_barrier(bar, k);
        }
    }
}

extern "C" void kernel_launch(void* const* d_in, const int* in_sizes, int n_in,
                              void* d_out, int out_size, void* d_ws, size_t ws_size,
                              hipStream_t stream)
{
    (void)in_sizes; (void)n_in; (void)out_size; (void)ws_size;
    const float* inX  = (const float*)d_in[0];
    const float* h0   = (const float*)d_in[1];
    const float* Wxz0 = (const float*)d_in[2];
    const float* bxz0 = (const float*)d_in[3];
    const float* Whz0 = (const float*)d_in[4];
    const float* Whr0 = (const float*)d_in[5];
    const float* Wxg0 = (const float*)d_in[6];
    const float* bxg0 = (const float*)d_in[7];
    const float* Whg0 = (const float*)d_in[8];
    const float* WxzS = (const float*)d_in[9];
    const float* bxzS = (const float*)d_in[10];
    const float* WhzS = (const float*)d_in[11];
    const float* WhrS = (const float*)d_in[12];
    const float* WxgS = (const float*)d_in[13];
    const float* bxgS = (const float*)d_in[14];
    const float* WhgS = (const float*)d_in[15];
    const float* Why  = (const float*)d_in[16];
    const float* bhy  = (const float*)d_in[17];

    char* ws = (char*)d_ws;
    int* bar = (int*)ws;
    size_t off = 4352;                                             // barrier counters (>= NTICKS ints)
    __bf16* wxg0b = (__bf16*)(ws + off); off += (size_t)512 * 128 * 2;
    __bf16* whg0b = (__bf16*)(ws + off); off += (size_t)512 * 512 * 2;
    __bf16* wxgSb = (__bf16*)(ws + off); off += (size_t)2 * 512 * 512 * 2;
    __bf16* whgSb = (__bf16*)(ws + off); off += (size_t)2 * 512 * 512 * 2;
    __bf16* whyb  = (__bf16*)(ws + off); off += (size_t)128 * 512 * 2;
    float*  hbuf  = (float*)(ws + off);  off += (size_t)NLAYER * 2 * BATCH * HDIM * 4;
    __bf16* xbuf  = (__bf16*)(ws + off); off += (size_t)NLAYER * 2 * BATCH * HDIM * 2;

    hipMemsetAsync(bar, 0, 4352, stream);
    f2bf_kernel<<<(65536 + 255) / 256, 256, 0, stream>>>(Wxg0, wxg0b, 65536);
    f2bf_kernel<<<(262144 + 255) / 256, 256, 0, stream>>>(Whg0, whg0b, 262144);
    f2bf_kernel<<<(524288 + 255) / 256, 256, 0, stream>>>(WxgS, wxgSb, 524288);
    f2bf_kernel<<<(524288 + 255) / 256, 256, 0, stream>>>(WhgS, whgSb, 524288);
    f2bf_kernel<<<(65536 + 255) / 256, 256, 0, stream>>>(Why, whyb, 65536);
    hinit_kernel<<<(BATCH * NLAYER * HDIM + 255) / 256, 256, 0, stream>>>(h0, hbuf);

    gru_persistent<<<TOTAL_WG, 512, 0, stream>>>(
        inX, Wxz0, bxz0, Whz0, Whr0, bxg0,
        WxzS, bxzS, WhzS, WhrS, bxgS, bhy,
        wxg0b, whg0b, wxgSb, whgSb, whyb,
        hbuf, xbuf, bar, (float*)d_out);
}
